// BlockToChannelPool_64355789963668
// MI455X (gfx1250) — matrix operand order repaired and checked by reading, not compiled, not run on hardware
//
#include <hip/hip_runtime.h>

// ---------- CDNA5 types ----------
typedef __bf16 bf16;
typedef bf16  v16bf __attribute__((ext_vector_type(16)));
typedef bf16  v8bf  __attribute__((ext_vector_type(8)));
typedef float v8f   __attribute__((ext_vector_type(8)));
typedef unsigned int v4u __attribute__((ext_vector_type(4)));
typedef int v8i __attribute__((ext_vector_type(8)));
typedef int v4i __attribute__((ext_vector_type(4)));

#define Bb 32
#define Nn 4096
#define Hh 256
#define Cc 96

// ---------- WMMA (wave32, 16x16x32 bf16 -> f32) ----------
__device__ __forceinline__ v8f wmma_bf16(v16bf a, v16bf b, v8f c) {
  return __builtin_amdgcn_wmma_f32_16x16x32_bf16(false, a, false, b, (short)0, c,
                                                 false, false);
}

// Fragment-major LDS staging: each 16x16(x32) fragment occupies 32 lanes x 16 bf16
// contiguously, so a consumer lane loads its whole v16bf with two ds_load_b128.
//
// A (16x32, MxK): lane = m + 16*((k>>3)&1); elem j = 8*(k>=16) + ((k&7)>>1)*2 + (k&1)
// B (32x16, KxN): lane = n + 16*((k>>3)&1); elem j = 8*(k>=16) + (k&7)
__device__ __forceinline__ void a_store2(bf16* frag, int m, int k /*even*/,
                                         float lo, float hi) {
  int lane = m + ((k & 8) ? 16 : 0);
  bf16* p = frag + lane * 16 + ((k & 16) ? 8 : 0) + ((k & 7) >> 1) * 2;
  p[0] = (bf16)lo;          // adjacent bytes -> v_cvt_pk_bf16_f32 + ds_store_b32
  p[1] = (bf16)hi;
}
__device__ __forceinline__ void b_store(bf16* frag, int k, int n, float v) {
  frag[(n + ((k & 8) ? 16 : 0)) * 16 + ((k & 16) ? 8 : 0) + (k & 7)] = (bf16)v;
}
__device__ __forceinline__ void b_store2(bf16* frag, int k /*even*/, int n,
                                         float lo, float hi) {
  bf16* p = frag + (n + ((k & 8) ? 16 : 0)) * 16 + ((k & 16) ? 8 : 0) + (k & 7);
  p[0] = (bf16)lo;
  p[1] = (bf16)hi;
}
__device__ __forceinline__ v16bf frag_load(const bf16* frags, int idx, int lane) {
  return *(const v16bf*)(frags + (idx * 32 + lane) * 16);   // 2x ds_load_b128
}

// D-matrix mapping: element r -> row r + 8*(lane>=16), col lane&15.
__device__ __forceinline__ int d_row(int r, int lane) { return r + ((lane >> 4) << 3); }
__device__ __forceinline__ int d_col(int lane) { return lane & 15; }

// Ordered-uint encoding so unsigned atomicMax == float max (handles -inf).
__device__ __forceinline__ unsigned fkey(float f) {
  unsigned u = __float_as_uint(f);
  return (u & 0x80000000u) ? ~u : (u | 0x80000000u);
}
__device__ __forceinline__ float funkey(unsigned k) {
  return __uint_as_float((k & 0x80000000u) ? (k & 0x7fffffffu) : ~k);
}

// Issue a 2D TDM load: tile_h rows x tile_w fp32 elements, row stride in elements.
// D# per CDNA5 ISA (group0: count/lds/global/type=2;
// group1: data_size=4B, dims, tile dims, dim0 stride).
__device__ __forceinline__ void tdm_load_2d(unsigned lds_off, const void* gptr,
                                            int tile_w, int tile_h, int row_stride) {
  unsigned long long ga = (unsigned long long)gptr;
  v4u g0;
  g0[0] = 1u;                                             // count = 1 descriptor
  g0[1] = lds_off;                                        // lds_addr
  g0[2] = (unsigned)(ga & 0xffffffffu);                   // global_addr[31:0]
  g0[3] = (unsigned)((ga >> 32) & 0x01ffffffu) | 0x80000000u; // addr[56:32]|type=2
  v8i g1;
  g1[0] = (2 << 16);                                      // data_size = 4 bytes
  g1[1] = (row_stride & 0xffff) << 16;                    // tensor_dim0[15:0]
  g1[2] = (row_stride >> 16) | (0x7fff << 16);            // dim0[31:16] | dim1[15:0]
  g1[3] = (tile_w << 16);                                 // dim1[31:16]=0 | tile_dim0
  g1[4] = tile_h;                                         // tile_dim1 | tile_dim2=0
  g1[5] = row_stride;                                     // tensor_dim0_stride[31:0]
  g1[6] = 0;
  g1[7] = 0;
  v4i gz = {0, 0, 0, 0};
#if defined(__clang_major__) && (__clang_major__ >= 23)
  v8i gz8 = {0, 0, 0, 0, 0, 0, 0, 0};
  __builtin_amdgcn_tensor_load_to_lds(g0, g1, gz, gz, gz8, 0);
#else
  __builtin_amdgcn_tensor_load_to_lds(g0, g1, gz, gz, 0);
#endif
}

// ---------------- Kernel 1: gate = tanh(x@W1+b1)@W2+b2, pad -> -inf ----------------
__global__ __launch_bounds__(256) void gate_kernel(
    const float* __restrict__ x, const float* __restrict__ W1, const float* __restrict__ b1,
    const float* __restrict__ W2, const float* __restrict__ b2,
    const unsigned char* __restrict__ pad, float* __restrict__ gate)
{
  __shared__ __attribute__((aligned(32))) bf16 sA[8 * 512];   // 8 A-frags (128 tokens)
  __shared__ __attribute__((aligned(32))) bf16 sB[8 * 512];   // 8 B-frags (128 hidden)
  __shared__ __attribute__((aligned(32))) bf16 sH[128 * 128]; // tanh staging
  const int tid = threadIdx.x, lane = tid & 31, wave = tid >> 5;
  const long long token0 = (long long)blockIdx.x * 128;

  v8f acc[8];
#pragma unroll
  for (int t = 0; t < 8; ++t)
#pragma unroll
    for (int e = 0; e < 8; ++e) acc[t][e] = 0.f;

  for (int k0 = 0; k0 < Hh; k0 += 32) {
    for (int p = tid; p < 2048; p += 256) {             // A: 128 rows x 16 k-pairs
      int row = p >> 4, kp = (p & 15) * 2;
      float2 v = *(const float2*)&x[(token0 + row) * Hh + k0 + kp];
      a_store2(sA + (row >> 4) * 512, row & 15, kp, v.x, v.y);
    }
    for (int i = tid; i < 4096; i += 256) {             // B: 32 k x 128 n (coalesced)
      int k = i >> 7, n = i & 127;
      b_store(sB + (n >> 4) * 512, k, n & 15, W1[(k0 + k) * 128 + n]);
    }
    __syncthreads();
    // Preload all fragments so the 18 ds_load_b128 issue back-to-back and one
    // dscnt wait covers the whole 8-WMMA burst (no per-WMMA LDS stall).
    v16bf a = frag_load(sA, wave, lane);
    v16bf bfr[8];
#pragma unroll
    for (int t = 0; t < 8; ++t) bfr[t] = frag_load(sB, t, lane);
#pragma unroll
    for (int t = 0; t < 8; ++t) acc[t] = wmma_bf16(a, bfr[t], acc[t]);
    __syncthreads();
  }

#pragma unroll
  for (int t = 0; t < 8; ++t) {
    int col = t * 16 + d_col(lane);
    float bias = b1[col];
#pragma unroll
    for (int r = 0; r < 8; ++r) {
      int row = wave * 16 + d_row(r, lane);
      sH[row * 128 + col] = (bf16)tanhf(acc[t][r] + bias);
    }
  }
  __syncthreads();

  if (tid < 128) {
    float s = b2[0];
#pragma unroll
    for (int k = 0; k < 128; k += 8) {
      v8bf h = *(const v8bf*)&sH[tid * 128 + k];        // ds_load_b128
#pragma unroll
      for (int e = 0; e < 8; ++e) s += (float)h[e] * W2[k + e];
    }
    long long n = token0 + tid;
    if (pad[n]) s = -__builtin_inff();
    gate[n] = s;
  }
}

// ---------------- Kernel 2: per-(b,c) softmax stats + has_ch + any_ch ----------------
__global__ __launch_bounds__(256) void stats_kernel(
    const float* __restrict__ gate, const int* __restrict__ ch,
    const unsigned char* __restrict__ pad,
    float* __restrict__ chmax, float* __restrict__ chsum,
    float* __restrict__ hasch_out, int* __restrict__ anych)
{
  __shared__ unsigned um[Cc];
  __shared__ float ssum[Cc];
  __shared__ int cnt[Cc];
  const int tid = threadIdx.x, b = blockIdx.x;
  if (tid < Cc) { um[tid] = 0u; ssum[tid] = 0.f; cnt[tid] = 0; }
  __syncthreads();
  for (int n = tid; n < Nn; n += 256) {
    long long i = (long long)b * Nn + n;
    if (!pad[i]) {
      int c = ch[i];
      atomicMax(&um[c], fkey(gate[i]));
      atomicAdd(&cnt[c], 1);
    }
  }
  __syncthreads();
  for (int n = tid; n < Nn; n += 256) {
    long long i = (long long)b * Nn + n;
    if (!pad[i]) {
      int c = ch[i];
      atomicAdd(&ssum[c], __expf(gate[i] - funkey(um[c])));
    }
  }
  __syncthreads();
  if (tid < Cc) {
    chmax[b * Cc + tid] = funkey(um[tid]);
    chsum[b * Cc + tid] = ssum[tid];
    int has = cnt[tid] > 0;
    hasch_out[b * Cc + tid] = has ? 1.f : 0.f;
    if (has) atomicOr(&anych[tid], 1);
  }
}

// ---------------- Kernel 3: pooled[b,c,:] = sum_n w[b,c,n] * x[b,n,:] ----------------
// grid (B, 2): 96 channels x 128 H-cols per block. Dense (96x32) weight tile is
// rebuilt per K-step; x tiles stream in via double-buffered TDM DMA.
__global__ __launch_bounds__(256) void pool_kernel(
    const float* __restrict__ x, const float* __restrict__ gate,
    const int* __restrict__ ch, const unsigned char* __restrict__ pad,
    const int* __restrict__ ctype, const float* __restrict__ emb,
    const float* __restrict__ chmax, const float* __restrict__ chsum,
    float* __restrict__ pooled)
{
  __shared__ __attribute__((aligned(32))) bf16 sW[6 * 512];   // 6 A-frags (96 channels)
  __shared__ __attribute__((aligned(32))) bf16 sX[8 * 512];   // 8 B-frags (128 H-cols)
  __shared__ __attribute__((aligned(16))) float sStage[2][32 * 128];
  __shared__ float sScale[Cc], sMax[Cc];
  const int tid = threadIdx.x, lane = tid & 31, wave = tid >> 5;
  const int b = blockIdx.x;
  const int col0 = blockIdx.y * 128;

  if (tid < Cc) {
    float s = chsum[b * Cc + tid];
    float m = 1.f + 0.1f * emb[ctype[b] * Cc + tid];
    sScale[tid] = (s > 0.f) ? (m / s) : 0.f;   // empty channel -> weights 0
    sMax[tid] = chmax[b * Cc + tid];
  }

  v8f acc[6];
#pragma unroll
  for (int t = 0; t < 6; ++t)
#pragma unroll
    for (int e = 0; e < 8; ++e) acc[t][e] = 0.f;

  if (wave == 0)
    tdm_load_2d((unsigned)(unsigned long long)&sStage[0][0],
                &x[((long long)b * Nn) * Hh + col0], 128, 32, Hh);
  __syncthreads();

  int nb = 0;
  for (int n0 = 0; n0 < Nn; n0 += 32) {
    for (int i = tid; i < 1536; i += 256) ((unsigned*)sW)[i] = 0u;  // zero weight tile
    if (wave == 0) {
      if (n0 + 32 < Nn) {
        tdm_load_2d((unsigned)(unsigned long long)&sStage[nb ^ 1][0],
                    &x[((long long)b * Nn + n0 + 32) * Hh + col0], 128, 32, Hh);
        __builtin_amdgcn_s_wait_tensorcnt(1);   // current buffer complete
      } else {
        __builtin_amdgcn_s_wait_tensorcnt(0);
      }
    }
    __syncthreads();

    if (tid < 32) {                              // scatter softmax weights
      long long n = (long long)b * Nn + n0 + tid;
      if (!pad[n]) {
        int c = ch[n];
        bf16 wv = (bf16)(__expf(gate[n] - sMax[c]) * sScale[c]);
        sW[((c & 15) + ((tid & 8) ? 16 : 0)) * 16 +
           ((tid & 16) ? 8 : 0) + ((tid & 7) >> 1) * 2 + (tid & 1) + (c >> 4) * 512] = wv;
      }
    }
    const float* stage = sStage[nb];
    for (int p = tid; p < 2048; p += 256) {      // convert fp32 stage -> bf16 B-frags
      int n = p & 127, kp = (p >> 7) * 2;
      b_store2(sX + (n >> 4) * 512, kp, n & 15, stage[kp * 128 + n],
               stage[(kp + 1) * 128 + n]);
    }
    __syncthreads();

    v16bf af[6], bfr[6];
#pragma unroll
    for (int i = 0; i < 6; ++i) {
      int t = wave * 6 + i;
      af[i]  = frag_load(sW, t >> 3, lane);
      bfr[i] = frag_load(sX, t & 7, lane);
    }
#pragma unroll
    for (int i = 0; i < 6; ++i) acc[i] = wmma_bf16(af[i], bfr[i], acc[i]);
    __syncthreads();
    nb ^= 1;
  }

#pragma unroll
  for (int i = 0; i < 6; ++i) {
    int t = wave * 6 + i;
    int col = col0 + (t & 7) * 16 + d_col(lane);
#pragma unroll
    for (int r = 0; r < 8; ++r) {
      int c = (t >> 3) * 16 + d_row(r, lane);
      pooled[((long long)b * Cc + c) * Hh + col] = acc[i][r];
    }
  }
}

// ---------------- Kernel 4: tokens = any_ch * ELU(LN(pooled@Wp + bp)) ----------------
__global__ __launch_bounds__(256) void proj_kernel(
    const float* __restrict__ pooled, const float* __restrict__ Wp, const float* __restrict__ bp,
    const float* __restrict__ ln_g, const float* __restrict__ ln_b,
    const int* __restrict__ anych, float* __restrict__ out)
{
  __shared__ __attribute__((aligned(32))) bf16 sA[4 * 512];
  __shared__ __attribute__((aligned(32))) bf16 sB[16 * 512];
  __shared__ __attribute__((aligned(32))) bf16 sY[64 * 256];
  __shared__ float mu[64], rs[64];
  const int tid = threadIdx.x, lane = tid & 31, wave = tid >> 5;
  const long long row0 = (long long)blockIdx.x * 64;
  const int rt = wave >> 1;

  v8f acc[8];
#pragma unroll
  for (int t = 0; t < 8; ++t)
#pragma unroll
    for (int e = 0; e < 8; ++e) acc[t][e] = 0.f;

  for (int k0 = 0; k0 < Hh; k0 += 32) {
    for (int p = tid; p < 1024; p += 256) {     // A: 64 rows x 16 k-pairs
      int row = p >> 4, kp = (p & 15) * 2;
      float2 v = *(const float2*)&pooled[(row0 + row) * Hh + k0 + kp];
      a_store2(sA + (row >> 4) * 512, row & 15, kp, v.x, v.y);
    }
    for (int i = tid; i < 8192; i += 256) {     // B: 32 k x 256 n
      int k = i >> 8, n = i & 255;
      b_store(sB + (n >> 4) * 512, k, n & 15, Wp[(k0 + k) * Hh + n]);
    }
    __syncthreads();
    v16bf a = frag_load(sA, rt, lane);
    v16bf bfr[8];
#pragma unroll
    for (int i = 0; i < 8; ++i) bfr[i] = frag_load(sB, (wave & 1) * 8 + i, lane);
#pragma unroll
    for (int i = 0; i < 8; ++i) acc[i] = wmma_bf16(a, bfr[i], acc[i]);
    __syncthreads();
  }

#pragma unroll
  for (int i = 0; i < 8; ++i) {
    int col = ((wave & 1) * 8 + i) * 16 + d_col(lane);
    float bias = bp[col];
#pragma unroll
    for (int r = 0; r < 8; ++r) {
      int row = rt * 16 + d_row(r, lane);
      sY[row * 256 + col] = (bf16)(acc[i][r] + bias);
    }
  }
  __syncthreads();

  if (tid < 64) {
    float s = 0.f, s2 = 0.f;
#pragma unroll
    for (int c = 0; c < 256; c += 8) {
      v8bf h = *(const v8bf*)&sY[tid * 256 + c];
#pragma unroll
      for (int e = 0; e < 8; ++e) { float v = (float)h[e]; s += v; s2 += v * v; }
    }
    float m = s * (1.f / 256.f);
    mu[tid] = m;
    rs[tid] = rsqrtf(s2 * (1.f / 256.f) - m * m + 1e-5f);
  }
  __syncthreads();

  for (int i = tid; i < 64 * 256; i += 256) {
    int r = i >> 8, c = i & 255;
    long long grow = row0 + r;
    float v = ((float)sY[i] - mu[r]) * rs[r] * ln_g[c] + ln_b[c];
    v = v > 0.f ? v : (__expf(v) - 1.f);        // ELU
    if (!anych[(int)(grow % Cc)]) v = 0.f;
    out[grow * Hh + c] = v;
  }
}

// ---------------- launcher ----------------
extern "C" void kernel_launch(void* const* d_in, const int* in_sizes, int n_in,
                              void* d_out, int out_size, void* d_ws, size_t ws_size,
                              hipStream_t stream) {
  (void)in_sizes; (void)n_in; (void)out_size; (void)ws_size;
  const float* x           = (const float*)d_in[0];
  const int* ctype         = (const int*)d_in[1];
  const int* ch            = (const int*)d_in[2];
  const unsigned char* pad = (const unsigned char*)d_in[3];  // jax bool -> 1 byte
  const float* W1          = (const float*)d_in[4];
  const float* b1          = (const float*)d_in[5];
  const float* W2          = (const float*)d_in[6];
  const float* b2          = (const float*)d_in[7];
  const float* emb         = (const float*)d_in[8];
  const float* Wp          = (const float*)d_in[9];
  const float* bp          = (const float*)d_in[10];
  const float* lng         = (const float*)d_in[11];
  const float* lnb         = (const float*)d_in[12];
  float* out = (float*)d_out;

  float* ws     = (float*)d_ws;
  float* gate   = ws;                        // B*N = 131072 f32
  float* chmax  = gate + Bb * Nn;            // 3072
  float* chsum  = chmax + Bb * Cc;           // 3072
  int*   anych  = (int*)(chsum + Bb * Cc);   // 96 i32 (padded to 128)
  float* pooled = (float*)(anych + 128);     // B*C*H = 786432 f32

  hipMemsetAsync(anych, 0, 128 * sizeof(int), stream);

  gate_kernel<<<(Bb * Nn) / 128, 256, 0, stream>>>(x, W1, b1, W2, b2, pad, gate);
  stats_kernel<<<Bb, 256, 0, stream>>>(gate, ch, pad, chmax, chsum,
                                       out + (long long)Bb * Cc * Hh, anych);
  pool_kernel<<<dim3(Bb, 2), 256, 0, stream>>>(x, gate, ch, pad, ctype, emb,
                                               chmax, chsum, pooled);
  proj_kernel<<<(Bb * Cc) / 64, 256, 0, stream>>>(pooled, Wp, bp, lng, lnb, anych, out);
}
